// my_AttLayer_4105988735234
// MI455X (gfx1250) — compile-verified, matrix operand order checked by reading, not run on hardware
//
#include <hip/hip_runtime.h>

typedef __attribute__((ext_vector_type(16))) __bf16 v16bf;
typedef __attribute__((ext_vector_type(8)))  float  v8f;
typedef unsigned short us;

#define B_   16
#define C_   512
#define HW_  1024
#define O_   256     // 8 caps * 32
#define NK_  8
#define KT_  41472   // 81 * 512
#define PW_  40      // 32 + 2*4 padded width

__device__ __forceinline__ us f2bf(float f) {
    unsigned int u = __float_as_uint(f);
    unsigned int r = u + 0x7FFFu + ((u >> 16) & 1u);
    return (us)(r >> 16);
}

// ---------------- fill / pack kernels ----------------

__global__ __launch_bounds__(256) void fill_zero_u32(unsigned int* p, int n) {
    int i = blockIdx.x * 256 + threadIdx.x;
    if (i < n) p[i] = 0u;
}

// x5 NCHW fp32 -> a1 [b][hw][c] bf16
__global__ __launch_bounds__(256) void pack_x5(const float* __restrict__ x5, us* __restrict__ a1) {
    int i = blockIdx.x * 256 + threadIdx.x;            // 8388608
    int c = i & 511; int rest = i >> 9;
    int hw = rest & 1023; int b = rest >> 10;
    a1[i] = f2bf(x5[(b * C_ + c) * HW_ + hw]);
}

// conv_w (o,c) fp32 -> bf16 (same layout)
__global__ __launch_bounds__(256) void pack_w1(const float* __restrict__ w, us* __restrict__ o) {
    int i = blockIdx.x * 256 + threadIdx.x;            // 262144
    o[i] = f2bf(w[i]);
}

// caps_w (o,c,kh,kw) fp32 -> wr[o][(kh*9+kw)*512 + c] bf16
__global__ __launch_bounds__(256) void pack_capsw(const float* __restrict__ w, us* __restrict__ wr) {
    int i = blockIdx.x * 256 + threadIdx.x;            // 10616832
    if (i >= O_ * KT_) return;
    int o = i / KT_;
    int rem = i - o * KT_;
    int r = rem >> 9;          // kh*9+kw
    int c = rem & 511;
    wr[i] = f2bf(w[(o * C_ + c) * 81 + r]);
}

// ---------------- WMMA GEMM 1: x = x5 @ W^T + b + x5 ----------------
// M=16384 N=512 K=512.  Wave tile 16x64 (4 accumulators).
__global__ __launch_bounds__(256)
void gemm1_wmma(const us* __restrict__ a1, const us* __restrict__ w1,
                const float* __restrict__ conv_b, const float* __restrict__ x5,
                float* __restrict__ x, us* __restrict__ xpad) {
    int tid = threadIdx.x, lane = tid & 31, wave = tid >> 5;
    int tile = blockIdx.x * 8 + wave;                  // 8192 tiles
    int tile_n = tile & 7, tile_m = tile >> 3;
    int m0 = tile_m * 16, n0 = tile_n * 64;
    int arow = (m0 + (lane & 15)) * C_;
    int klad = (lane >> 4) << 4;
    int wb[4];
#pragma unroll
    for (int nf = 0; nf < 4; ++nf) wb[nf] = (n0 + nf * 16 + (lane & 15)) * C_;

    v8f z = {};
    v8f acc[4];
#pragma unroll
    for (int nf = 0; nf < 4; ++nf) acc[nf] = z;

    for (int kk = 0; kk < 16; ++kk) {
        int k = kk * 32 + klad;
        v16bf av = *(const v16bf*)(a1 + arow + k);
#pragma unroll
        for (int nf = 0; nf < 4; ++nf) {
            v16bf bv = *(const v16bf*)(w1 + wb[nf] + k);
            acc[nf] = __builtin_amdgcn_wmma_f32_16x16x32_bf16(
                false, av, false, bv, (short)0, acc[nf], false, false);
        }
    }
    int mhalf = (lane >> 4) << 3;
#pragma unroll
    for (int nf = 0; nf < 4; ++nf) {
        int n = n0 + nf * 16 + (lane & 15);
        float bias = conv_b[n];
#pragma unroll
        for (int v = 0; v < 8; ++v) {
            int m = m0 + v + mhalf;
            int b = m >> 10, hw = m & 1023;
            int h = hw >> 5, w = hw & 31;
            int xi = (b * C_ + n) * HW_ + hw;
            float val = acc[nf][v] + bias + x5[xi];
            x[xi] = val;
            xpad[((b * PW_ + h + 4) * PW_ + (w + 4)) * C_ + n] = f2bf(val);
        }
    }
}

// ---------------- WMMA implicit-GEMM 9x9 conv, LDS double-buffered ----------------
// M=16384 N=256 K=41472.
// Workgroup macro-tile: 128(M) x 128(N); 8 waves in 4(M) x 2(N) grid, 32x64 per wave.
// K staged in 64-wide slabs through LDS (double-buffered, 1 barrier / slab).
#define ROWUS 80   // LDS row stride in us units: 64k*2B data + 32B pad = 160B (32B aligned)

__global__ __launch_bounds__(256)
void conv9x9_wmma(const us* __restrict__ xpad, const us* __restrict__ wr,
                  const float* __restrict__ caps_b, float* __restrict__ cap) {
    __shared__ __align__(32) us Abuf[2][128 * ROWUS];   // 2 x 20 KB
    __shared__ __align__(32) us Bbuf[2][128 * ROWUS];   // 2 x 20 KB
    __shared__ int offs[81];

    int tid = threadIdx.x, lane = tid & 31, wave = tid >> 5;
    if (tid < 81) { int kh = tid / 9, kw = tid - kh * 9; offs[tid] = (kh * PW_ + kw) * C_; }

    int Mblk = blockIdx.x >> 1;                // 0..127
    int Nblk = blockIdx.x & 1;                 // 0..1
    int m0 = Mblk * 128;

    // ---- staging assignment: thread handles A-chunks {tid, tid+256} and same for B.
    // chunk id c: row = c>>2 (0..127), k-half16 = c&3
    int row0 = tid >> 2;                       // 0..63 ; second chunk row0+64
    int ahalf = tid & 3;
    int aB0, aB1;
    {
        int m = m0 + row0;
        int b = m >> 10, hw = m & 1023, h = hw >> 5, w = hw & 31;
        aB0 = ((b * PW_ + h) * PW_ + w) * C_;
        m = m0 + row0 + 64;
        b = m >> 10; hw = m & 1023; h = hw >> 5; w = hw & 31;
        aB1 = ((b * PW_ + h) * PW_ + w) * C_;
    }
    int nB0 = (Nblk * 128 + row0) * KT_;
    int nB1 = nB0 + 64 * KT_;
    int s0 = row0 * ROWUS + ahalf * 16;        // LDS (us units)
    int s1 = s0 + 64 * ROWUS;

    // ---- compute assignment
    int wm = (wave >> 1) * 32;                 // wave M origin within macro-tile
    int wn = (wave & 1) * 64;                  // wave N origin within macro-tile
    int l15 = lane & 15;
    int lk = (lane >> 4) << 4;                 // 0 or 16 (us units in k)

    v8f z = {};
    v8f acc[2][4];
#pragma unroll
    for (int mf = 0; mf < 2; ++mf)
#pragma unroll
        for (int nf = 0; nf < 4; ++nf) acc[mf][nf] = z;

    v16bf sa0, sa1, sb0, sb1;
    __syncthreads();                           // offs[] ready

    // prologue: stage slab 0 into buffer 0
    {
        int kbase = 0;
        int ao = offs[0] + ahalf * 16;
        sa0 = *(const v16bf*)(xpad + aB0 + ao);
        sa1 = *(const v16bf*)(xpad + aB1 + ao);
        int ko = kbase + ahalf * 16;
        sb0 = *(const v16bf*)(wr + nB0 + ko);
        sb1 = *(const v16bf*)(wr + nB1 + ko);
        *(v16bf*)&Abuf[0][s0] = sa0;
        *(v16bf*)&Abuf[0][s1] = sa1;
        *(v16bf*)&Bbuf[0][s0] = sb0;
        *(v16bf*)&Bbuf[0][s1] = sb1;
    }

    int buf = 0;
    for (int it = 0; it < 648; ++it) {
        __syncthreads();                       // buf[it&1] fully staged
        bool pf = (it + 1) < 648;
        if (pf) {                              // issue next slab's global loads early
            int kbase = (it + 1) * 64;
            int r = kbase >> 9;
            int ao = offs[r] + (kbase & 511) + ahalf * 16;
            sa0 = *(const v16bf*)(xpad + aB0 + ao);
            sa1 = *(const v16bf*)(xpad + aB1 + ao);
            int ko = kbase + ahalf * 16;
            sb0 = *(const v16bf*)(wr + nB0 + ko);
            sb1 = *(const v16bf*)(wr + nB1 + ko);
        }
        // compute two k-steps (K=32 each) from LDS
#pragma unroll
        for (int ks = 0; ks < 2; ++ks) {
            int kofs = ks * 32 + lk;           // us units
            v16bf bfr[4];
#pragma unroll
            for (int nf = 0; nf < 4; ++nf)
                bfr[nf] = *(const v16bf*)&Bbuf[buf][(wn + nf * 16 + l15) * ROWUS + kofs];
#pragma unroll
            for (int mf = 0; mf < 2; ++mf) {
                v16bf afr = *(const v16bf*)&Abuf[buf][(wm + mf * 16 + l15) * ROWUS + kofs];
#pragma unroll
                for (int nf = 0; nf < 4; ++nf)
                    acc[mf][nf] = __builtin_amdgcn_wmma_f32_16x16x32_bf16(
                        false, afr, false, bfr[nf], (short)0, acc[mf][nf], false, false);
            }
        }
        if (pf) {
            int nb = buf ^ 1;
            *(v16bf*)&Abuf[nb][s0] = sa0;
            *(v16bf*)&Abuf[nb][s1] = sa1;
            *(v16bf*)&Bbuf[nb][s0] = sb0;
            *(v16bf*)&Bbuf[nb][s1] = sb1;
        }
        buf ^= 1;
    }

    // epilogue
    int mh8 = (lane >> 4) << 3;
#pragma unroll
    for (int mf = 0; mf < 2; ++mf) {
#pragma unroll
        for (int nf = 0; nf < 4; ++nf) {
            int n = Nblk * 128 + wn + nf * 16 + l15;
            float bias = caps_b[n];
#pragma unroll
            for (int v = 0; v < 8; ++v) {
                int m = m0 + wm + mf * 16 + v + mh8;
                int b = m >> 10, hw = m & 1023;
                cap[(b * O_ + n) * HW_ + hw] = acc[mf][nf][v] + bias;
            }
        }
    }
}

// ---------------- squash + per-capsule spatial-feature max -> cap4 ----------------
__global__ __launch_bounds__(256)
void squash_max_k(const float* __restrict__ cap, float* __restrict__ cap4) {
    int g = blockIdx.x * 256 + threadIdx.x;            // (b,hw), 16384
    int b = g >> 10, hw = g & 1023;
    float mx[NK_];
#pragma unroll
    for (int k = 0; k < NK_; ++k) mx[k] = -3.4e38f;
    for (int cc = 0; cc < 32; ++cc) {
        float t[NK_];
        float sq = 0.f;
#pragma unroll
        for (int k = 0; k < NK_; ++k) {
            t[k] = cap[(b * O_ + (k * 32 + cc)) * HW_ + hw];
            sq += t[k] * t[k];
        }
        float f = sq / ((1.f + sq) * sqrtf(sq + 1e-8f));
#pragma unroll
        for (int k = 0; k < NK_; ++k) mx[k] = fmaxf(mx[k], t[k] * f);
    }
#pragma unroll
    for (int k = 0; k < NK_; ++k) cap4[g * NK_ + k] = mx[k];
}

// per-(b,k) spatial max of cap4
__global__ __launch_bounds__(256)
void rowmax_k(const float* __restrict__ cap4, float* __restrict__ Mmax) {
    __shared__ float red[256];
    int bk = blockIdx.x;                               // 128
    int b = bk >> 3, k = bk & 7;
    float m = -3.4e38f;
    for (int hw = threadIdx.x; hw < HW_; hw += 256)
        m = fmaxf(m, cap4[(b * HW_ + hw) * NK_ + k]);
    red[threadIdx.x] = m;
    __syncthreads();
    for (int s = 128; s > 0; s >>= 1) {
        if (threadIdx.x < s) red[threadIdx.x] = fmaxf(red[threadIdx.x], red[threadIdx.x + s]);
        __syncthreads();
    }
    if (threadIdx.x == 0) Mmax[bk] = red[0];
}

// per-(b,hw) 1/max(||x||,eps) over channels
__global__ __launch_bounds__(256)
void invnorm_k(const float* __restrict__ x, float* __restrict__ invn) {
    int g = blockIdx.x * 256 + threadIdx.x;            // 16384
    int b = g >> 10, hw = g & 1023;
    float s = 0.f;
    for (int c = 0; c < C_; ++c) {
        float v = x[(b * C_ + c) * HW_ + hw];
        s += v * v;
    }
    invn[g] = 1.f / fmaxf(sqrtf(s), 1e-12f);
}

// seeds[b,c,k] = sum over matched hw of norm0[b,c,hw]   (mask == top-1 incl. ties)
__global__ __launch_bounds__(256)
void seeds_k(const float* __restrict__ cap4, const float* __restrict__ Mmax,
             const float* __restrict__ invn, const float* __restrict__ x,
             float* __restrict__ seeds) {
    __shared__ int cnt;
    __shared__ int idxs[1024];
    __shared__ float wts[1024];
    int bk = blockIdx.x;                               // 128
    int b = bk >> 3, k = bk & 7;
    float mv = Mmax[bk];
    if (threadIdx.x == 0) cnt = 0;
    __syncthreads();
    for (int hw = threadIdx.x; hw < HW_; hw += 256) {
        if (cap4[(b * HW_ + hw) * NK_ + k] == mv) {
            int p = atomicAdd(&cnt, 1);
            idxs[p] = hw;
            wts[p] = invn[b * HW_ + hw];
        }
    }
    __syncthreads();
    int nmatch = cnt;
    for (int c = threadIdx.x; c < C_; c += 256) {
        float a = 0.f;
        for (int j = 0; j < nmatch; ++j)
            a += x[(b * C_ + c) * HW_ + idxs[j]] * wts[j];
        seeds[(b * C_ + c) * NK_ + k] = a;
    }
}

__global__ __launch_bounds__(256)
void smean_k(const float* __restrict__ seeds, float* __restrict__ smean) {
    int i = blockIdx.x * 256 + threadIdx.x;            // 4096 = c*8+k
    if (i >= C_ * NK_) return;
    float s = 0.f;
    for (int b = 0; b < B_; ++b) s += seeds[b * C_ * NK_ + i];
    smean[i] = s * (1.f / B_);
}

// cm[b,k,hw] = invn[b,hw] * sum_c x[b,c,hw] * smean[c,k]
__global__ __launch_bounds__(256)
void cm_k(const float* __restrict__ x, const float* __restrict__ invn,
          const float* __restrict__ smean, float* __restrict__ cm) {
    __shared__ float sm[C_ * NK_];
    for (int i = threadIdx.x; i < C_ * NK_; i += 256) sm[i] = smean[i];
    __syncthreads();
    int g = blockIdx.x * 256 + threadIdx.x;            // 16384
    int b = g >> 10, hw = g & 1023;
    float acc[NK_];
#pragma unroll
    for (int k = 0; k < NK_; ++k) acc[k] = 0.f;
    for (int c = 0; c < C_; ++c) {
        float xv = x[(b * C_ + c) * HW_ + hw];
#pragma unroll
        for (int k = 0; k < NK_; ++k) acc[k] += xv * sm[c * NK_ + k];
    }
    float iv = invn[g];
#pragma unroll
    for (int k = 0; k < NK_; ++k) cm[(b * NK_ + k) * HW_ + hw] = acc[k] * iv;
}

__global__ __launch_bounds__(256)
void minmax_k(const float* __restrict__ cm, float* __restrict__ mn, float* __restrict__ mx) {
    __shared__ float rmn[256], rmx[256];
    int bk = blockIdx.x;                               // 128
    float lo = 3.4e38f, hi = -3.4e38f;
    for (int hw = threadIdx.x; hw < HW_; hw += 256) {
        float v = cm[bk * HW_ + hw];
        lo = fminf(lo, v); hi = fmaxf(hi, v);
    }
    rmn[threadIdx.x] = lo; rmx[threadIdx.x] = hi;
    __syncthreads();
    for (int s = 128; s > 0; s >>= 1) {
        if (threadIdx.x < s) {
            rmn[threadIdx.x] = fminf(rmn[threadIdx.x], rmn[threadIdx.x + s]);
            rmx[threadIdx.x] = fmaxf(rmx[threadIdx.x], rmx[threadIdx.x + s]);
        }
        __syncthreads();
    }
    if (threadIdx.x == 0) { mn[bk] = rmn[0]; mx[bk] = rmx[0]; }
}

__global__ __launch_bounds__(256)
void cmnorm_k(float* __restrict__ cm, const float* __restrict__ mn, const float* __restrict__ mx) {
    int i = blockIdx.x * 256 + threadIdx.x;            // 131072
    int bk = i >> 10;
    cm[i] = (cm[i] - mn[bk]) / (mx[bk] - mn[bk] + 1e-12f);
}

// proto[k,c] = mean over (b,hw) of x*cm
__global__ __launch_bounds__(256)
void proto_k(const float* __restrict__ x, const float* __restrict__ cm,
             float* __restrict__ proto) {
    __shared__ float red[256];
    int blk = blockIdx.x;                              // 4096 = k*512+c
    int k = blk >> 9, c = blk & 511;
    float s = 0.f;
    for (int i = threadIdx.x; i < B_ * HW_; i += 256) {
        int b = i >> 10, hw = i & 1023;
        s += x[(b * C_ + c) * HW_ + hw] * cm[(b * NK_ + k) * HW_ + hw];
    }
    red[threadIdx.x] = s;
    __syncthreads();
    for (int t = 128; t > 0; t >>= 1) {
        if (threadIdx.x < t) red[threadIdx.x] += red[threadIdx.x + t];
        __syncthreads();
    }
    if (threadIdx.x == 0) proto[k * C_ + c] = red[0] * (1.f / (B_ * HW_));
}

__global__ __launch_bounds__(256)
void pmean_k(const float* __restrict__ proto, float* __restrict__ pmean) {
    int c = blockIdx.x * 256 + threadIdx.x;            // 512
    if (c >= C_) return;
    float s = 0.f;
    for (int k = 0; k < NK_; ++k) s += proto[k * C_ + c];
    pmean[c] = s * (1.f / NK_);
}

__global__ __launch_bounds__(256)
void cmmean_k(const float* __restrict__ cm, float* __restrict__ cmm) {
    int g = blockIdx.x * 256 + threadIdx.x;            // 16384
    int b = g >> 10, hw = g & 1023;
    float s = 0.f;
    for (int k = 0; k < NK_; ++k) s += cm[(b * NK_ + k) * HW_ + hw];
    cmm[g] = s * (1.f / NK_);
}

__global__ __launch_bounds__(512)
void outpro_k(const float* __restrict__ pmean, float* __restrict__ out) {
    __shared__ float red[512];
    float v = pmean[threadIdx.x];
    red[threadIdx.x] = v * v;
    __syncthreads();
    for (int s = 256; s > 0; s >>= 1) {
        if (threadIdx.x < s) red[threadIdx.x] += red[threadIdx.x + s];
        __syncthreads();
    }
    float inv = 1.f / fmaxf(sqrtf(red[0]), 1e-12f);
    out[threadIdx.x] = v * inv;
}

__global__ __launch_bounds__(256)
void outmap_k(const float* __restrict__ x, const float* __restrict__ pmean,
              const float* __restrict__ cmm, float* __restrict__ out) {
    __shared__ float pm[C_];
    for (int i = threadIdx.x; i < C_; i += 256) pm[i] = pmean[i];
    __syncthreads();
    int g = blockIdx.x * 256 + threadIdx.x;            // 16384
    int b = g >> 10, hw = g & 1023;
    float cmv = cmm[g];
    float s = 0.f;
    for (int c = 0; c < C_; ++c) {
        float v = x[(b * C_ + c) * HW_ + hw] * (pm[c] + cmv);
        s += v * v;
    }
    float inv = 1.f / fmaxf(sqrtf(s), 1e-12f);
    for (int c = 0; c < C_; ++c) {
        float v = x[(b * C_ + c) * HW_ + hw] * (pm[c] + cmv);
        out[(b * C_ + c) * HW_ + hw] = v * inv;
    }
}

// ---------------- host launcher ----------------
extern "C" void kernel_launch(void* const* d_in, const int* in_sizes, int n_in,
                              void* d_out, int out_size, void* d_ws, size_t ws_size,
                              hipStream_t stream) {
    const float* x5     = (const float*)d_in[0];
    const float* conv_w = (const float*)d_in[1];
    const float* conv_b = (const float*)d_in[2];
    const float* caps_w = (const float*)d_in[3];
    const float* caps_b = (const float*)d_in[4];

    char* p = (char*)d_ws;
    auto alloc = [&](size_t bytes) -> void* {
        void* r = (void*)p;
        p += (bytes + 255) & ~(size_t)255;
        return r;
    };
    float* x     = (float*)alloc((size_t)B_ * C_ * HW_ * 4);        // 32 MB
    us*    xpad  = (us*)   alloc((size_t)B_ * PW_ * PW_ * C_ * 2);  // 25.6 MB
    us*    a1    = (us*)   alloc((size_t)B_ * HW_ * C_ * 2);        // 16 MB
    us*    w1    = (us*)   alloc((size_t)C_ * C_ * 2);
    us*    wr    = (us*)   alloc((size_t)O_ * KT_ * 2);             // 21.2 MB
    float* cap   = (float*)alloc((size_t)B_ * O_ * HW_ * 4);        // 16 MB
    float* cap4  = (float*)alloc((size_t)B_ * HW_ * NK_ * 4);
    float* Mmax  = (float*)alloc(B_ * NK_ * 4);
    float* invn  = (float*)alloc((size_t)B_ * HW_ * 4);
    float* seeds = (float*)alloc((size_t)B_ * C_ * NK_ * 4);
    float* smean = (float*)alloc((size_t)C_ * NK_ * 4);
    float* cm    = (float*)alloc((size_t)B_ * NK_ * HW_ * 4);
    float* mn    = (float*)alloc(B_ * NK_ * 4);
    float* mx    = (float*)alloc(B_ * NK_ * 4);
    float* proto = (float*)alloc((size_t)NK_ * C_ * 4);
    float* pmean = (float*)alloc(C_ * 4);
    float* cmm   = (float*)alloc((size_t)B_ * HW_ * 4);

    float* out_pro = (float*)d_out;           // 512
    float* out_map = (float*)d_out + C_;      // 16*512*1024

    // zero halo of padded activations
    {
        int nd = (B_ * PW_ * PW_ * C_ * 2) / 4;   // 6,553,600 dwords
        fill_zero_u32<<<(nd + 255) / 256, 256, 0, stream>>>((unsigned int*)xpad, nd);
    }
    pack_x5<<<(B_ * HW_ * C_) / 256, 256, 0, stream>>>(x5, a1);
    pack_w1<<<(C_ * C_) / 256, 256, 0, stream>>>(conv_w, w1);
    pack_capsw<<<(O_ * KT_ + 255) / 256, 256, 0, stream>>>(caps_w, wr);

    gemm1_wmma<<<1024, 256, 0, stream>>>(a1, w1, conv_b, x5, x, xpad);
    conv9x9_wmma<<<256, 256, 0, stream>>>(xpad, wr, caps_b, cap);

    squash_max_k<<<64, 256, 0, stream>>>(cap, cap4);
    rowmax_k<<<B_ * NK_, 256, 0, stream>>>(cap4, Mmax);
    invnorm_k<<<64, 256, 0, stream>>>(x, invn);
    seeds_k<<<B_ * NK_, 256, 0, stream>>>(cap4, Mmax, invn, x, seeds);
    smean_k<<<(C_ * NK_ + 255) / 256, 256, 0, stream>>>(seeds, smean);
    cm_k<<<64, 256, 0, stream>>>(x, invn, smean, cm);
    minmax_k<<<B_ * NK_, 256, 0, stream>>>(cm, mn, mx);
    cmnorm_k<<<(B_ * NK_ * HW_) / 256, 256, 0, stream>>>(cm, mn, mx);
    proto_k<<<NK_ * C_, 256, 0, stream>>>(x, cm, proto);
    pmean_k<<<(C_ + 255) / 256, 256, 0, stream>>>(proto, pmean);
    cmmean_k<<<64, 256, 0, stream>>>(cm, cmm);
    outpro_k<<<1, 512, 0, stream>>>(pmean, out_pro);
    outmap_k<<<64, 256, 0, stream>>>(x, pmean, cmm, out_map);
}
